// Mamba_65420941852740
// MI455X (gfx1250) — matrix operand order repaired and checked by reading, not compiled, not run on hardware
//
#include <hip/hip_runtime.h>
#include <hip/hip_bf16.h>

// ---------------------------------------------------------------------------
// Mamba-style block for MI455X (gfx1250, wave32, WMMA + TDM).
//   1) k_rmsnorm         : x -> xn (bf16 K-pair packed)
//   2) k_pack_w (x2)     : Win/Wout -> K-pair packed bf16
//   3) k_gemm_bf16       : u = xn @ Win + b_in        (v_wmma_f32_16x16x32_bf16,
//                          TDM tensor_load_to_lds double-buffered staging)
//   4) k_matpow          : T[d] = A[d]^128            (v_wmma_f32_16x16x4_f32)
//   5) k_scan_pass<0>    : per-chunk partial states p_c (zero init, conv fused)
//   6) k_combine         : s0_{c+1} = s0_c @ T + p_c
//   7) k_scan_pass<1>    : re-run chunks from s0_c, emit y (bf16)
//   8) k_gemm_bf16       : out = y @ Wout + b_out + x
// ---------------------------------------------------------------------------

#define D_MODEL 1024
#define D_IN    2048
#define D_STATE 16
#define CONV_D  4
#define BATCH   4
#define SEQ     2048
#define MROWS   (BATCH * SEQ)   // 8192

#define CHUNKS   16
#define CHUNK_L  (SEQ / CHUNKS)   // 128
#define MP_ITERS 7                // 2^7 = 128 = CHUNK_L

typedef __attribute__((ext_vector_type(16))) __bf16        v16bf;
typedef __attribute__((ext_vector_type(8)))  float         v8f;
typedef __attribute__((ext_vector_type(2)))  float         v2f;
typedef __attribute__((ext_vector_type(4)))  unsigned int  v4u;
typedef __attribute__((ext_vector_type(8)))  int           v8i;
typedef __attribute__((ext_vector_type(4)))  int           v4i;

#if defined(__gfx1250__) && __has_builtin(__builtin_amdgcn_tensor_load_to_lds)
#define USE_TDM 1
#else
#define USE_TDM 0
#endif

__device__ __forceinline__ unsigned short f2bf(float f) {
    unsigned int u = __float_as_uint(f);
    u += 0x7FFFu + ((u >> 16) & 1u);      // round-to-nearest-even
    return (unsigned short)(u >> 16);
}
__device__ __forceinline__ unsigned int pack2bf(float lo, float hi) {
    return (unsigned int)f2bf(lo) | ((unsigned int)f2bf(hi) << 16);
}

// ---------------------------------------------------------------------------
// 1) RMSNorm
// ---------------------------------------------------------------------------
__global__ __launch_bounds__(256) void k_rmsnorm(const float* __restrict__ x,
                                                 const float* __restrict__ w,
                                                 unsigned int* __restrict__ xnpk) {
    __shared__ float red[256];
    const int row = blockIdx.x;
    const int tid = threadIdx.x;
    const float4 v = ((const float4*)(x + (size_t)row * D_MODEL))[tid];
    red[tid] = v.x * v.x + v.y * v.y + v.z * v.z + v.w * v.w;
    __syncthreads();
    for (int s = 128; s > 0; s >>= 1) {
        if (tid < s) red[tid] += red[tid + s];
        __syncthreads();
    }
    const float rstd = rsqrtf(red[0] * (1.0f / D_MODEL) + 1e-6f);
    const float4 wv = ((const float4*)w)[tid];
    unsigned int* dst = xnpk + (size_t)row * (D_MODEL / 2) + tid * 2;
    dst[0] = pack2bf(v.x * rstd * wv.x, v.y * rstd * wv.y);
    dst[1] = pack2bf(v.z * rstd * wv.z, v.w * rstd * wv.w);
}

// ---------------------------------------------------------------------------
// 2) Pack W[K][N] f32 -> Wpk[kp][n] u32 = (bf16 W[2kp][n], bf16 W[2kp+1][n]).
// ---------------------------------------------------------------------------
__global__ __launch_bounds__(256) void k_pack_w(const float* __restrict__ W,
                                                unsigned int* __restrict__ Wpk,
                                                int N, long total) {
    long idx = (long)blockIdx.x * 256 + threadIdx.x;
    if (idx >= total) return;
    long kp = idx / N;
    long n  = idx - kp * N;
    Wpk[idx] = pack2bf(W[(2 * kp) * (long)N + n], W[(2 * kp + 1) * (long)N + n]);
}

// ---------------------------------------------------------------------------
// 3/8) bf16 WMMA GEMM, 128x128 block tile, 8 waves (32x64 each), K-step 32.
//      Staging: TDM (tensor_load_to_lds) double-buffered, with hardware LDS
//      padding matching the bank-conflict-free strides; fallback = register
//      double-buffering.
// ---------------------------------------------------------------------------
#define GBM 128
#define GBN 128
#define AS_STRIDE 20    // u32 per A row in LDS (16 data + 4 pad)
#define BS_STRIDE 132   // u32 per B kpair-row in LDS (128 data + 4 pad)

__device__ __forceinline__ v8f wmma_bf16(v16bf a, v16bf b, v8f c) {
    return __builtin_amdgcn_wmma_f32_16x16x32_bf16(false, a, false, b,
                                                   (short)0, c, false, false);
}
__device__ __forceinline__ v8f wmma_f32(v2f a, v2f b, v8f c) {
    return __builtin_amdgcn_wmma_f32_16x16x4_f32(false, a, false, b,
                                                 (short)0, c, false, false);
}

#if USE_TDM
// 2-D TDM load: tile (tile_d1 rows x tile_d0 u32) from a row-major u32 tensor
// (row length stride_d0, in elements) into LDS at lds_off, padding pad_amt
// DWORDs after every 2^(pad_int_code+1) DWORDs.  ISA 8.3/8.4 descriptor.
__device__ __forceinline__ void tdm_load_2d(unsigned int lds_off,
                                            const void* gaddr,
                                            unsigned int tensor_d0,
                                            unsigned int tensor_d1,
                                            unsigned int tile_d0,
                                            unsigned int tile_d1,
                                            unsigned int stride_d0,
                                            unsigned int pad_int_code,
                                            unsigned int pad_amt_code) {
    const unsigned long long ga = (unsigned long long)gaddr;
    v4u g0;
    g0[0] = 1u;                                          // count=1, user mode
    g0[1] = lds_off;                                     // lds_addr (bytes)
    g0[2] = (unsigned int)(ga & 0xFFFFFFFFull);          // global_addr[31:0]
    g0[3] = (unsigned int)((ga >> 32) & 0x1FFFFFFull)    // global_addr[56:32]
            | (2u << 30);                                // type=2 ("image")
    v8i g1;
    g1[0] = (int)((2u << 16)                              // data_size = 4B
                  | (1u << 20)                            // pad_enable
                  | (pad_int_code << 22)
                  | (pad_amt_code << 25));                // workgroup_mask=0
    g1[1] = (int)((tensor_d0 & 0xFFFFu) << 16);           // tensor_dim0 lo16
    g1[2] = (int)(((tensor_d0 >> 16) & 0xFFFFu)
                  | ((tensor_d1 & 0xFFFFu) << 16));       // dim0 hi / dim1 lo
    g1[3] = (int)(((tensor_d1 >> 16) & 0xFFFFu)
                  | ((tile_d0 & 0xFFFFu) << 16));         // dim1 hi / tile_dim0
    g1[4] = (int)(tile_d1 & 0xFFFFu);                     // tile_dim1, tile_dim2=0
    g1[5] = (int)stride_d0;                               // tensor_dim0_stride lo32
    g1[6] = 0;                                            // stride hi16, dim1_stride lo16
    g1[7] = 0;
    const v4i z = {0, 0, 0, 0};
#if __clang_major__ >= 23
    const v8i z8 = {0, 0, 0, 0, 0, 0, 0, 0};
    __builtin_amdgcn_tensor_load_to_lds(g0, g1, z, z, z8, 0);
#else
    __builtin_amdgcn_tensor_load_to_lds(g0, g1, z, z, 0);
#endif
}
__device__ __forceinline__ unsigned int lds_offset_of(const void* p) {
    return (unsigned int)(unsigned long long)p;   // LDS aperture lives in [63:32]
}
#endif

// A fragment (16x32 bf16): lane M = l%16, h = l>=16; v0..3 <- kpairs 4h..4h+3,
// v4..7 <- kpairs 8+4h..11+4h.
__device__ __forceinline__ v16bf load_frag_a(const unsigned int* As, int lane,
                                             int mBase) {
    const unsigned int* row = As + (mBase + (lane & 15)) * AS_STRIDE;
    const int h4 = (lane >> 4) * 4;
    uint4 lo = *(const uint4*)(row + h4);
    uint4 hi = *(const uint4*)(row + 8 + h4);
    union { unsigned int u[8]; v16bf v; } r;
    r.u[0] = lo.x; r.u[1] = lo.y; r.u[2] = lo.z; r.u[3] = lo.w;
    r.u[4] = hi.x; r.u[5] = hi.y; r.u[6] = hi.z; r.u[7] = hi.w;
    return r.v;
}

// B fragment (32x16 bf16): lane N = l%16, h = l>=16; v0..3 <- kpairs 4h..4h+3,
// v4..7 <- kpairs 8+4h..11+4h.
__device__ __forceinline__ v16bf load_frag_b(const unsigned int* Bs, int lane,
                                             int nBase) {
    const int n  = nBase + (lane & 15);
    const int kh = (lane >> 4) * 4;
    union { unsigned int u[8]; v16bf v; } r;
#pragma unroll
    for (int v = 0; v < 8; ++v) r.u[v] = Bs[(kh + (v & 3) + ((v >> 2) << 3)) * BS_STRIDE + n];
    return r.v;
}

__global__ __launch_bounds__(256) void k_gemm_bf16(
    const unsigned int* __restrict__ Apk,   // [M][K/2] u32 pairs
    const unsigned int* __restrict__ Bpk,   // [K/2][N] u32 pairs
    const float* __restrict__ bias,         // [N]
    const float* __restrict__ residual,     // [M][N] or nullptr
    float* __restrict__ C,                  // [M][N]
    int M, int N, int K) {
    const int tid  = threadIdx.x;
    const int wave = tid >> 5;
    const int lane = tid & 31;
    const int wm   = wave >> 1;   // 0..3
    const int wn   = wave & 1;    // 0..1
    const int tileM = blockIdx.y * GBM;
    const int tileN = blockIdx.x * GBN;
    const int Kp = K >> 1;
    const int nk = Kp >> 4;

    v8f acc[2][4] = {};

#if USE_TDM
    __shared__ __align__(16) unsigned int As[2][GBM * AS_STRIDE];
    __shared__ __align__(16) unsigned int Bs[2][16 * BS_STRIDE];

    const bool w0 = (wave == 0);
    // prologue: DMA tile 0 into buffer 0
    if (w0) {
        tdm_load_2d(lds_offset_of(&As[0][0]), Apk + (size_t)tileM * Kp,
                    Kp, M, 16, GBM, Kp, /*pad every 16 dw*/ 3, /*+4 dw*/ 3);
        tdm_load_2d(lds_offset_of(&Bs[0][0]), Bpk + tileN,
                    N, Kp, 128, 16, N, /*pad every 128 dw*/ 6, /*+4 dw*/ 3);
    }
    for (int kt = 0; kt < nk; ++kt) {
        __syncthreads();   // prior compute done -> alt buffer reusable
        if (w0) {
            if (kt + 1 < nk) {
                const int bn = (kt + 1) & 1;
                tdm_load_2d(lds_offset_of(&As[bn][0]),
                            Apk + (size_t)tileM * Kp + (kt + 1) * 16,
                            Kp, M, 16, GBM, Kp, 3, 3);
                tdm_load_2d(lds_offset_of(&Bs[bn][0]),
                            Bpk + (size_t)((kt + 1) * 16) * N + tileN,
                            N, Kp, 128, 16, N, 6, 3);
                __builtin_amdgcn_s_wait_tensorcnt(2);  // tile kt resident
            } else {
                __builtin_amdgcn_s_wait_tensorcnt(0);
            }
        }
        __syncthreads();   // release all waves onto buffer kt&1
        const unsigned int* Ac = &As[kt & 1][0];
        const unsigned int* Bc = &Bs[kt & 1][0];
        v16bf a0 = load_frag_a(Ac, lane, wm * 32);
        v16bf a1 = load_frag_a(Ac, lane, wm * 32 + 16);
#pragma unroll
        for (int ni = 0; ni < 4; ++ni) {
            v16bf b = load_frag_b(Bc, lane, wn * 64 + ni * 16);
            acc[0][ni] = wmma_bf16(a0, b, acc[0][ni]);
            acc[1][ni] = wmma_bf16(a1, b, acc[1][ni]);
        }
    }
#else
    __shared__ __align__(16) unsigned int As[1][GBM * AS_STRIDE];
    __shared__ __align__(16) unsigned int Bs[1][16 * BS_STRIDE];
    unsigned int ra[8], rb[8];
#pragma unroll
    for (int r = 0; r < 8; ++r) {
        int e = tid + 256 * r;
        ra[r] = Apk[(size_t)(tileM + (e >> 4)) * Kp + (e & 15)];
        rb[r] = Bpk[(size_t)(e >> 7) * N + tileN + (e & 127)];
    }
    for (int kt = 0; kt < nk; ++kt) {
#pragma unroll
        for (int r = 0; r < 8; ++r) {
            int e = tid + 256 * r;
            As[0][(e >> 4) * AS_STRIDE + (e & 15)] = ra[r];
            Bs[0][(e >> 7) * BS_STRIDE + (e & 127)] = rb[r];
        }
        __syncthreads();
        if (kt + 1 < nk) {
#pragma unroll
            for (int r = 0; r < 8; ++r) {
                int e = tid + 256 * r;
                ra[r] = Apk[(size_t)(tileM + (e >> 4)) * Kp + (kt + 1) * 16 + (e & 15)];
                rb[r] = Bpk[(size_t)((kt + 1) * 16 + (e >> 7)) * N + tileN + (e & 127)];
            }
        }
        v16bf a0 = load_frag_a(&As[0][0], lane, wm * 32);
        v16bf a1 = load_frag_a(&As[0][0], lane, wm * 32 + 16);
#pragma unroll
        for (int ni = 0; ni < 4; ++ni) {
            v16bf b = load_frag_b(&Bs[0][0], lane, wn * 64 + ni * 16);
            acc[0][ni] = wmma_bf16(a0, b, acc[0][ni]);
            acc[1][ni] = wmma_bf16(a1, b, acc[1][ni]);
        }
        __syncthreads();
    }
#endif

    // ---- branch-hoisted epilogue; C/D layout: M = r + 8*(l>=16), N = l%16 ----
    const int lm = (lane >> 4) * 8;
    const int ln = lane & 15;
    float bias_r[4];
#pragma unroll
    for (int ni = 0; ni < 4; ++ni)
        bias_r[ni] = bias[tileN + wn * 64 + ni * 16 + ln];

    if (residual != nullptr) {
#pragma unroll
        for (int mi = 0; mi < 2; ++mi)
#pragma unroll
            for (int r = 0; r < 8; ++r) {
                const size_t rowb =
                    (size_t)(tileM + wm * 32 + mi * 16 + r + lm) * N;
#pragma unroll
                for (int ni = 0; ni < 4; ++ni) {
                    const size_t idx = rowb + tileN + wn * 64 + ni * 16 + ln;
                    C[idx] = acc[mi][ni][r] + bias_r[ni] + residual[idx];
                }
            }
    } else {
#pragma unroll
        for (int mi = 0; mi < 2; ++mi)
#pragma unroll
            for (int r = 0; r < 8; ++r) {
                const size_t rowb =
                    (size_t)(tileM + wm * 32 + mi * 16 + r + lm) * N;
#pragma unroll
                for (int ni = 0; ni < 4; ++ni) {
                    const size_t idx = rowb + tileN + wn * 64 + ni * 16 + ln;
                    C[idx] = acc[mi][ni][r] + bias_r[ni];
                }
            }
    }
}

// ---------------------------------------------------------------------------
// 4) T[d] = A[d]^CHUNK_L via repeated squaring with f32 WMMA (16x16x4 x4).
// ---------------------------------------------------------------------------
#define MP_WAVES 4
__global__ __launch_bounds__(128) void k_matpow(const float* __restrict__ A,
                                                float* __restrict__ T) {
    __shared__ __align__(16) float Ms[MP_WAVES][16 * 17];
    const int lane = threadIdx.x & 31;
    const int w    = threadIdx.x >> 5;
    const int d    = blockIdx.x * MP_WAVES + w;
    float* Mm = &Ms[w][0];

#pragma unroll
    for (int e0 = 0; e0 < 256; e0 += 32) {
        int e = e0 + lane;
        Mm[(e >> 4) * 17 + (e & 15)] = A[(size_t)d * 256 + e];
    }
    const int m = lane & 15;
    const int h2 = (lane >> 4) * 2;

    for (int it = 0; it < MP_ITERS; ++it) {
        v8f acc = {};
#pragma unroll
        for (int kk = 0; kk < 4; ++kk) {
            v2f a, b;
            a.x = Mm[m * 17 + kk * 4 + h2];
            a.y = Mm[m * 17 + kk * 4 + h2 + 1];
            b.x = Mm[(kk * 4 + h2) * 17 + m];
            b.y = Mm[(kk * 4 + h2 + 1) * 17 + m];
            acc = wmma_f32(a, b, acc);
        }
#pragma unroll
        for (int r = 0; r < 8; ++r)
            Mm[(r + 4 * h2) * 17 + m] = acc[r];   // same-wave DS in-order: WAR safe
    }
#pragma unroll
    for (int e0 = 0; e0 < 256; e0 += 32) {
        int e = e0 + lane;
        T[(size_t)d * 256 + e] = Mm[(e >> 4) * 17 + (e & 15)];
    }
}

// ---------------------------------------------------------------------------
// 5/7) Chunked conv+scan. Block = 32 channels x 4 batches; grid (D_IN/32, C).
// ---------------------------------------------------------------------------
#define SCH 32
#define A_LDS_STRIDE 260

template <bool EMIT>
__global__ __launch_bounds__(128) void k_scan_pass(
    const float* __restrict__ u,       // [B*SEQ][D_IN]
    const float* __restrict__ A,       // [D_IN][16][16]
    const float* __restrict__ Bm,      // [D_IN][16]
    const float* __restrict__ Cm,      // [D_IN][16]
    const float* __restrict__ convw,   // [D_IN][4]
    const float* __restrict__ convb,   // [D_IN]
    const float* __restrict__ ss0,     // [C][B][D_IN][16]  (EMIT only)
    float* __restrict__ pout,          // [C][B][D_IN][16]  (!EMIT only)
    unsigned short* __restrict__ ybf)  // [B*SEQ][D_IN] bf16 (EMIT only)
{
    __shared__ __align__(16) float Al[SCH * A_LDS_STRIDE];
    const int tid = threadIdx.x;
    const int b   = tid & 3;
    const int ch  = tid >> 2;
    const int d   = blockIdx.x * SCH + ch;
    const int c   = blockIdx.y;
    const int t0  = c * CHUNK_L;

#pragma unroll 4
    for (int q = 0; q < 64; ++q)
        Al[ch * A_LDS_STRIDE + b * 64 + q] = A[(size_t)d * 256 + b * 64 + q];
    __syncthreads();

    float Bv[16], Cv[16];
#pragma unroll
    for (int i = 0; i < 16; ++i) Bv[i] = Bm[d * 16 + i];
    if (EMIT) {
#pragma unroll
        for (int i = 0; i < 16; ++i) Cv[i] = Cm[d * 16 + i];
    }
    const float w0 = convw[d * 4 + 0], w1 = convw[d * 4 + 1];
    const float w2 = convw[d * 4 + 2], w3 = convw[d * 4 + 3];
    const float cb = convb[d];

    const size_t sbase = ((size_t)(c * BATCH + b) * D_IN + d) * 16;
    float s[16];
    if (EMIT) {
#pragma unroll
        for (int j = 0; j < 16; ++j) s[j] = ss0[sbase + j];
    } else {
#pragma unroll
        for (int j = 0; j < 16; ++j) s[j] = 0.0f;
    }

    const size_t urow = (size_t)b * SEQ * D_IN + d;
    float u1 = (t0 >= 1) ? u[urow + (size_t)(t0 - 1) * D_IN] : 0.0f;
    float u2 = (t0 >= 2) ? u[urow + (size_t)(t0 - 2) * D_IN] : 0.0f;
    float u3 = (t0 >= 3) ? u[urow + (size_t)(t0 - 3) * D_IN] : 0.0f;

    const float* Ach = &Al[ch * A_LDS_STRIDE];

    for (int t = t0; t < t0 + CHUNK_L; ++t) {
        const float u0 = u[urow + (size_t)t * D_IN];
        const float uc = w0 * u3 + w1 * u2 + w2 * u1 + w3 * u0 + cb;
        u3 = u2; u2 = u1; u1 = u0;

        float sn[16];
#pragma unroll
        for (int j = 0; j < 16; ++j) sn[j] = uc * Bv[j];
#pragma unroll
        for (int i = 0; i < 16; ++i) {
            const float si = s[i];
#pragma unroll
            for (int j = 0; j < 16; j += 4) {
                const float4 a = *(const float4*)&Ach[i * 16 + j];
                sn[j + 0] += si * a.x;
                sn[j + 1] += si * a.y;
                sn[j + 2] += si * a.z;
                sn[j + 3] += si * a.w;
            }
        }
        if (EMIT) {
            float y = 0.0f;
#pragma unroll
            for (int j = 0; j < 16; ++j) {
                y += sn[j] * Cv[j];
                s[j] = sn[j];
            }
            ybf[urow + (size_t)t * D_IN] = f2bf(y);
        } else {
#pragma unroll
            for (int j = 0; j < 16; ++j) s[j] = sn[j];
        }
    }
    if (!EMIT) {
#pragma unroll
        for (int j = 0; j < 16; ++j) pout[sbase + j] = s[j];
    }
}

// ---------------------------------------------------------------------------
// 6) Combine: s0_0 = 0; s0_{c+1} = s0_c @ T[d] + p_c.
// ---------------------------------------------------------------------------
__global__ __launch_bounds__(128) void k_combine(
    const float* __restrict__ T,      // [D_IN][16][16]
    const float* __restrict__ p,      // [C][B][D_IN][16]
    float* __restrict__ ss0)          // [C][B][D_IN][16]
{
    __shared__ __align__(16) float Tl[SCH * A_LDS_STRIDE];
    const int tid = threadIdx.x;
    const int b   = tid & 3;
    const int ch  = tid >> 2;
    const int d   = blockIdx.x * SCH + ch;

#pragma unroll 4
    for (int q = 0; q < 64; ++q)
        Tl[ch * A_LDS_STRIDE + b * 64 + q] = T[(size_t)d * 256 + b * 64 + q];
    __syncthreads();

    const float* Tch = &Tl[ch * A_LDS_STRIDE];
    float s[16];
#pragma unroll
    for (int j = 0; j < 16; ++j) s[j] = 0.0f;

    for (int c = 0; c < CHUNKS; ++c) {
        const size_t base = ((size_t)(c * BATCH + b) * D_IN + d) * 16;
#pragma unroll
        for (int j = 0; j < 16; ++j) ss0[base + j] = s[j];

        float sn[16];
#pragma unroll
        for (int j = 0; j < 16; ++j) sn[j] = p[base + j];
#pragma unroll
        for (int i = 0; i < 16; ++i) {
            const float si = s[i];
#pragma unroll
            for (int j = 0; j < 16; j += 4) {
                const float4 a = *(const float4*)&Tch[i * 16 + j];
                sn[j + 0] += si * a.x;
                sn[j + 1] += si * a.y;
                sn[j + 2] += si * a.z;
                sn[j + 3] += si * a.w;
            }
        }
#pragma unroll
        for (int j = 0; j < 16; ++j) s[j] = sn[j];
    }
}

// ---------------------------------------------------------------------------
// Launch
// ---------------------------------------------------------------------------
extern "C" void kernel_launch(void* const* d_in, const int* in_sizes, int n_in,
                              void* d_out, int out_size, void* d_ws, size_t ws_size,
                              hipStream_t stream) {
    const float* x      = (const float*)d_in[0];
    const float* norm_w = (const float*)d_in[1];
    const float* Win    = (const float*)d_in[2];
    const float* b_in   = (const float*)d_in[3];
    const float* conv_w = (const float*)d_in[4];
    const float* conv_b = (const float*)d_in[5];
    const float* A      = (const float*)d_in[6];
    const float* Bm     = (const float*)d_in[7];
    const float* Cm     = (const float*)d_in[8];
    const float* Wout   = (const float*)d_in[9];
    const float* b_out  = (const float*)d_in[10];
    float* out = (float*)d_out;

    char* ws = (char*)d_ws;
    size_t off = 0;
    auto wsalloc = [&](size_t bytes) -> void* {
        void* p = ws + off;
        off += (bytes + 255) & ~(size_t)255;
        return p;
    };
    unsigned int* xn    = (unsigned int*)wsalloc((size_t)MROWS * (D_MODEL / 2) * 4); // 16 MB
    unsigned int* WinP  = (unsigned int*)wsalloc((size_t)(D_MODEL / 2) * D_IN * 4);  //  4 MB
    unsigned int* WoutP = (unsigned int*)wsalloc((size_t)(D_IN / 2) * D_MODEL * 4);  //  4 MB
    float*        u     = (float*)       wsalloc((size_t)MROWS * D_IN * 4);          // 64 MB
    unsigned int* ybf   = (unsigned int*)wsalloc((size_t)MROWS * D_IN * 2);          // 32 MB
    float*        Tm    = (float*)       wsalloc((size_t)D_IN * 256 * 4);            //  2 MB
    float*        pbuf  = (float*)       wsalloc((size_t)CHUNKS * BATCH * D_IN * 16 * 4); // 8 MB
    float*        s0buf = (float*)       wsalloc((size_t)CHUNKS * BATCH * D_IN * 16 * 4); // 8 MB

    k_rmsnorm<<<MROWS, 256, 0, stream>>>(x, norm_w, xn);

    {
        long tot1 = (long)(D_MODEL / 2) * D_IN;
        long tot2 = (long)(D_IN / 2) * D_MODEL;
        k_pack_w<<<(unsigned)((tot1 + 255) / 256), 256, 0, stream>>>(Win, WinP, D_IN, tot1);
        k_pack_w<<<(unsigned)((tot2 + 255) / 256), 256, 0, stream>>>(Wout, WoutP, D_MODEL, tot2);
    }

    {
        dim3 grid(D_IN / GBN, MROWS / GBM);
        k_gemm_bf16<<<grid, 256, 0, stream>>>(xn, WinP, b_in, nullptr, u,
                                              MROWS, D_IN, D_MODEL);
    }

    k_matpow<<<D_IN / MP_WAVES, 128, 0, stream>>>(A, Tm);

    {
        dim3 grid(D_IN / SCH, CHUNKS);
        k_scan_pass<false><<<grid, 128, 0, stream>>>(u, A, Bm, Cm, conv_w, conv_b,
                                                     nullptr, pbuf, nullptr);
    }

    k_combine<<<D_IN / SCH, 128, 0, stream>>>(Tm, pbuf, s0buf);

    {
        dim3 grid(D_IN / SCH, CHUNKS);
        k_scan_pass<true><<<grid, 128, 0, stream>>>(u, A, Bm, Cm, conv_w, conv_b,
                                                    s0buf, nullptr,
                                                    (unsigned short*)ybf);
    }

    {
        dim3 grid(D_MODEL / GBN, MROWS / GBM);
        k_gemm_bf16<<<grid, 256, 0, stream>>>(ybf, WoutP, b_out, x, out,
                                              MROWS, D_MODEL, D_IN);
    }
}
